// CRF_37211596652577
// MI455X (gfx1250) — matrix operand order repaired and checked by reading, not compiled, not run on hardware
//
#include <hip/hip_runtime.h>

// CRF log-likelihood, B=256, S=512, T=128 (PAD_ID=127).
// Forward recursion recast as exp-domain GEMM:
//   s'[b,j] = m[b] + ln( sum_i exp(s[b,i]-m[b]) * P[i,j] ) + em[t,b,j],  P = exp(trans)
// Per step, per 16-batch tile: (16x128 f16) @ (128x128 f16) via v_wmma_f32_16x16x32_f16,
// B fragments resident in VGPRs for all 512 steps.

typedef __attribute__((ext_vector_type(16))) _Float16 v16h;
typedef __attribute__((ext_vector_type(8)))  _Float16 v8h;
typedef __attribute__((ext_vector_type(8)))  float    v8f;
typedef __attribute__((ext_vector_type(4)))  float    f32x4;

constexpr int NTAGS  = 128;
constexpr int PAD_ID = 127;
constexpr int SEQ    = 512;    // S (fixed by reference; enables constant IOFFSET addressing)
constexpr int NB     = 256;    // B
constexpr int SP     = 132;    // s_lds row stride (floats): 528B = 33*16, bank-staggered
constexpr int AP     = 136;    // a_lds row stride (halves): 272B = 17*16, bank-staggered
constexpr float LOG2E = 1.4426950408889634f;
constexpr float LN2   = 0.6931471805599453f;

// ---------------------------------------------------------------- prep: P = exp(trans) in f16
__global__ __launch_bounds__(256) void crf_prep_kernel(const float* __restrict__ trans,
                                                       _Float16* __restrict__ Pexp) {
  int i = blockIdx.x * 256 + threadIdx.x;
  if (i < NTAGS * NTAGS) Pexp[i] = (_Float16)__builtin_amdgcn_exp2f(trans[i] * LOG2E);
}

// ---------------------------------------------------------------- numerator (gather path)
__global__ __launch_bounds__(256) void crf_num_kernel(const float* __restrict__ em,
                                                      const int*   __restrict__ tags,
                                                      const float* __restrict__ start_t,
                                                      const float* __restrict__ end_t,
                                                      const float* __restrict__ trans,
                                                      float* __restrict__ num) {
  int b = blockIdx.x * blockDim.x + threadIdx.x;
  if (b >= NB) return;
  const int*   tg = tags + (size_t)b * SEQ;
  const float* eb = em   + (size_t)b * SEQ * NTAGS;
  int prev = tg[0];
  float score = start_t[prev] + eb[prev];
  int cnt = (prev != PAD_ID) ? 1 : 0;
  for (int t = 1; t < SEQ; ++t) {
    int cur = tg[t];
    if (cur != PAD_ID) {
      score += trans[prev * NTAGS + cur] + eb[(size_t)t * NTAGS + cur];
      ++cnt;
    }
    prev = cur;
  }
  int lastPos = cnt > 0 ? cnt - 1 : 0;
  num[b] = score + end_t[tg[lastPos]];
}

// ---------------------------------------------------------------- forward algorithm (WMMA)
__global__ __launch_bounds__(256) void crf_forward_kernel(const float* __restrict__ em,
                                                          const int*   __restrict__ tags,
                                                          const float* __restrict__ start_t,
                                                          const float* __restrict__ end_t,
                                                          const _Float16* __restrict__ Pexp,
                                                          float* __restrict__ den) {
  __shared__ __align__(16) float    s_lds[16 * SP];   // running log-alpha: 16 batches x 128 tags
  __shared__ __align__(16) _Float16 a_lds[16 * AP];   // exp(s - m) in f16
  __shared__ __align__(16) float    m_lds[16];
  __shared__ __align__(16) float    maskf_lds[16];
  __shared__ float red[16][17];

  const int tid   = threadIdx.x;
  const int lane  = tid & 31;
  const int wave  = tid >> 5;                // N-block 0..7 (cols 16*wave .. +16)
  const int bBase = blockIdx.x * 16;

  const int ncol  = wave * 16 + (lane & 15); // this lane's output column (tag j)
  const int hi8   = (lane >> 4) * 8;         // A-fragment K sub-offset, upper half-wave
  const int mbase = (lane >> 4) * 8;         // C/D row base, upper half-wave

  const int rb    = tid >> 4;                // reduction phase: batch row (0..15)
  const int j0    = (tid & 15) * 8;          // reduction phase: 8-tag chunk

  // Per-lane emissions base: row (bBase+mbase), column ncol, step 0.
  // Per step, 8 loads at constant IOFFSETs r*SEQ*NTAGS*4 (<= 1.75MB, fits 24-bit).
  const float* em_lane = em + ((size_t)(bBase + mbase) * SEQ) * NTAGS + ncol;

  // B fragments (exp(trans) tile for this wave's 16 columns), resident for all steps.
  // B layout used here (contiguous K, same as A): lane holds N = lane&15; element e -> K = (lane>=16)*16 + e.
  v16h bfrag[4];
#pragma unroll
  for (int kb = 0; kb < 4; ++kb) {
    int kbase = kb * 32 + (lane >> 4) * 16;
#pragma unroll
    for (int e = 0; e < 16; ++e)
      bfrag[kb][e] = Pexp[(kbase + e) * NTAGS + ncol];
  }

  // s0 = start + em[0]
  {
    const float* e0 = em + (size_t)(bBase + rb) * SEQ * NTAGS;
#pragma unroll
    for (int q = 0; q < 8; ++q) {
      int j = j0 + q;
      s_lds[rb * SP + j] = start_t[j] + e0[j];
    }
  }
  __syncthreads();

  for (int t = 1; t < SEQ; ++t) {
    // ---- issue this step's global loads early (overlap with max/exp/WMMA) ----
    float emv[8];
#pragma unroll
    for (int r = 0; r < 8; ++r)
      emv[r] = em_lane[(size_t)r * SEQ * NTAGS + (size_t)t * NTAGS];
    int tagv = 0;
    if (tid < 16) tagv = tags[(size_t)(bBase + tid) * SEQ + t];

    // ---- per-batch-row max: 8 local + half-wave shuffle reduction (no barrier) ----
    f32x4 s0 = *(const f32x4*)(s_lds + rb * SP + j0);
    f32x4 s1 = *(const f32x4*)(s_lds + rb * SP + j0 + 4);
    float mx = fmaxf(fmaxf(fmaxf(s0.x, s0.y), fmaxf(s0.z, s0.w)),
                     fmaxf(fmaxf(s1.x, s1.y), fmaxf(s1.z, s1.w)));
    mx = fmaxf(mx, __shfl_xor(mx, 1, 16));
    mx = fmaxf(mx, __shfl_xor(mx, 2, 16));
    mx = fmaxf(mx, __shfl_xor(mx, 4, 16));
    mx = fmaxf(mx, __shfl_xor(mx, 8, 16));

    // ---- a = exp2((s-m)*log2e) -> f16, one ds_store_b128 per thread ----
    {
      v8h ah;
      ah[0] = (_Float16)__builtin_amdgcn_exp2f((s0.x - mx) * LOG2E);
      ah[1] = (_Float16)__builtin_amdgcn_exp2f((s0.y - mx) * LOG2E);
      ah[2] = (_Float16)__builtin_amdgcn_exp2f((s0.z - mx) * LOG2E);
      ah[3] = (_Float16)__builtin_amdgcn_exp2f((s0.w - mx) * LOG2E);
      ah[4] = (_Float16)__builtin_amdgcn_exp2f((s1.x - mx) * LOG2E);
      ah[5] = (_Float16)__builtin_amdgcn_exp2f((s1.y - mx) * LOG2E);
      ah[6] = (_Float16)__builtin_amdgcn_exp2f((s1.z - mx) * LOG2E);
      ah[7] = (_Float16)__builtin_amdgcn_exp2f((s1.w - mx) * LOG2E);
      *(v8h*)(a_lds + rb * AP + j0) = ah;
    }
    if ((tid & 15) == 0) m_lds[rb] = mx;
    if (tid < 16) maskf_lds[tid] = (tagv != PAD_ID) ? 1.0f : 0.0f;
    __syncthreads();   // #1: a_lds/m_lds/maskf ready; all s_lds reads done

    // prefetch next step's emissions (global_prefetch_b8)
    if (t + 1 < SEQ)
      __builtin_prefetch(&em_lane[(size_t)(lane & 7) * SEQ * NTAGS + (size_t)(t + 1) * NTAGS], 0, 1);

    // ---- z = a @ P : batch all 8 A-fragment ds_load_b128, then 4 chained WMMAs ----
    union { v16h v; v8h h[2]; } af[4];
    {
      const _Float16* arow = a_lds + (lane & 15) * AP;
#pragma unroll
      for (int kb = 0; kb < 4; ++kb) {
        af[kb].h[0] = *(const v8h*)(arow + kb * 32 + 2 * hi8);       // e=0..7  : K = kb*32+2*hi8+e
        af[kb].h[1] = *(const v8h*)(arow + kb * 32 + 2 * hi8 + 8);  // e=8..15 : K = kb*32+2*hi8+e
      }
    }
    v8f acc = {};
#pragma unroll
    for (int kb = 0; kb < 4; ++kb)
      acc = __builtin_amdgcn_wmma_f32_16x16x32_f16(false, af[kb].v, false, bfrag[kb],
                                                   (short)0, acc, false, false);

    // ---- branchless masked update: s = old + maskf*(m + ln2*log2(z) + em - old) ----
    f32x4 mr0 = *(const f32x4*)(m_lds + mbase);
    f32x4 mr1 = *(const f32x4*)(m_lds + mbase + 4);
    f32x4 mf0 = *(const f32x4*)(maskf_lds + mbase);
    f32x4 mf1 = *(const f32x4*)(maskf_lds + mbase + 4);
    float mrow[8] = {mr0.x, mr0.y, mr0.z, mr0.w, mr1.x, mr1.y, mr1.z, mr1.w};
    float mfr[8]  = {mf0.x, mf0.y, mf0.z, mf0.w, mf1.x, mf1.y, mf1.z, mf1.w};
#pragma unroll
    for (int r = 0; r < 8; ++r) {
      float* sp   = s_lds + (mbase + r) * SP + ncol;
      float oldv  = *sp;
      float newv  = fmaf(LN2, __builtin_amdgcn_logf(acc[r]), mrow[r] + emv[r]);
      *sp = fmaf(mfr[r], newv - oldv, oldv);
    }
    __syncthreads();   // #2
  }

  // ---- denominator = logsumexp_j(s + end) ----
  {
    float mx = -3.0e38f;
#pragma unroll
    for (int q = 0; q < 8; ++q) {
      int j = j0 + q;
      mx = fmaxf(mx, s_lds[rb * SP + j] + end_t[j]);
    }
    red[rb][tid & 15] = mx;
  }
  __syncthreads();
  if (tid < 16) {
    float mx = red[tid][0];
#pragma unroll
    for (int q = 1; q < 16; ++q) mx = fmaxf(mx, red[tid][q]);
    m_lds[tid] = mx;
  }
  __syncthreads();
  {
    float m = m_lds[rb], sm = 0.f;
#pragma unroll
    for (int q = 0; q < 8; ++q) {
      int j = j0 + q;
      sm += __builtin_amdgcn_exp2f((s_lds[rb * SP + j] + end_t[j] - m) * LOG2E);
    }
    red[rb][tid & 15] = sm;
  }
  __syncthreads();
  if (tid < 16) {
    float sm = 0.f;
#pragma unroll
    for (int q = 0; q < 16; ++q) sm += red[tid][q];
    den[bBase + tid] = fmaf(LN2, __builtin_amdgcn_logf(sm), m_lds[tid]);
  }
}

// ---------------------------------------------------------------- mean(num - den)
__global__ __launch_bounds__(256) void crf_reduce_kernel(const float* __restrict__ num,
                                                         const float* __restrict__ den,
                                                         float* __restrict__ out) {
  __shared__ float sh[256];
  int tid = threadIdx.x;
  sh[tid] = (tid < NB) ? (num[tid] - den[tid]) : 0.f;
  __syncthreads();
  for (int s = 128; s > 0; s >>= 1) {
    if (tid < s) sh[tid] += sh[tid + s];
    __syncthreads();
  }
  if (tid == 0) out[0] = sh[0] / (float)NB;
}

extern "C" void kernel_launch(void* const* d_in, const int* in_sizes, int n_in,
                              void* d_out, int out_size, void* d_ws, size_t ws_size,
                              hipStream_t stream) {
  const float* emissions = (const float*)d_in[0];   // [B,S,T] f32
  const int*   tags      = (const int*)  d_in[1];   // [B,S]
  const float* start_t   = (const float*)d_in[2];   // [T]
  const float* end_t     = (const float*)d_in[3];   // [T]
  const float* trans     = (const float*)d_in[4];   // [T,T]

  _Float16* Pexp = (_Float16*)d_ws;                            // 128*128 f16 = 32KB
  float*    den  = (float*)((char*)d_ws + sizeof(_Float16) * NTAGS * NTAGS);
  float*    num  = den + NB;

  crf_prep_kernel<<<(NTAGS * NTAGS + 255) / 256, 256, 0, stream>>>(trans, Pexp);
  crf_num_kernel<<<1, 256, 0, stream>>>(emissions, tags, start_t, end_t, trans, num);
  crf_forward_kernel<<<NB / 16, 256, 0, stream>>>(emissions, tags, start_t, end_t, Pexp, den);
  crf_reduce_kernel<<<1, 256, 0, stream>>>(num, den, (float*)d_out);
}
